// LSTMsBlock_1632087572934
// MI455X (gfx1250) — matrix operand order repaired and checked by reading, not compiled, not run on hardware
//
#include <hip/hip_runtime.h>

// Problem constants (from reference): T=512, B=64, D=512, H=1024, 4H=4096, L=2
#define TT 512
#define BB 64
#define DD 512
#define HH 1024
#define GG 4096
#define LL 2

typedef _Float16 half8  __attribute__((ext_vector_type(8)));
typedef _Float16 half16 __attribute__((ext_vector_type(16)));
typedef float    float8 __attribute__((ext_vector_type(8)));

union V16 { half16 v; half8 h[2]; };

// A-fragment (16x32 f16, MxK): lane row = lane&15; kb = (lane<16)?0:8.
// halves [0..7] = K kb..kb+7 (one b128), halves [8..15] = K 16+kb..16+kb+7 (one b128)
__device__ __forceinline__ half16 load_a_frag(const _Float16* __restrict__ rowp, int kb) {
  V16 u;
  u.h[0] = *(const half8*)(rowp + kb);
  u.h[1] = *(const half8*)(rowp + 16 + kb);
  return u.v;
}

// B-fragment (32x16 f16, KxN) with B stored transposed [N,K]:
// lane col = lane&15; same K order as A: halves [0..7] = K kb..kb+7, [8..15] = K 16+kb..16+kb+7 (kb = kg/2)
__device__ __forceinline__ half16 load_b_frag(const _Float16* __restrict__ rowp, int kg) {
  V16 u;
  u.h[0] = *(const half8*)(rowp + (kg >> 1));
  u.h[1] = *(const half8*)(rowp + 16 + (kg >> 1));
  return u.v;
}

__device__ __forceinline__ float sigmoidf_(float x) {
  return 1.0f / (1.0f + __expf(-x));
}

#define WMMA_F16(a, b, c) \
  __builtin_amdgcn_wmma_f32_16x16x32_f16(false, (a), false, (b), (short)0, (c), false, false)

// Software-pipelined GEMM segment: acc[mt] += A[mt-tile rows, LEN] x B[LEN, 16].
// Next iteration's 5 fragments are loaded BEFORE the current 4 WMMAs issue, so the
// WMMAs only wait on loads issued one full iteration earlier (no loadcnt-0 stalls).
template <int STRIDE, int LEN>
__device__ __forceinline__ void gemm_seg(const _Float16* __restrict__ arow0,
                                         const _Float16* __restrict__ brow,
                                         int kb, int kg, float8 (&acc)[4]) {
  half16 bc  = load_b_frag(brow, kg);
  half16 ac0 = load_a_frag(arow0 + (size_t)0 * 16 * STRIDE, kb);
  half16 ac1 = load_a_frag(arow0 + (size_t)1 * 16 * STRIDE, kb);
  half16 ac2 = load_a_frag(arow0 + (size_t)2 * 16 * STRIDE, kb);
  half16 ac3 = load_a_frag(arow0 + (size_t)3 * 16 * STRIDE, kb);
  #pragma unroll 2
  for (int k = 32; k < LEN; k += 32) {
    const half16 bn  = load_b_frag(brow + k, kg);
    const half16 an0 = load_a_frag(arow0 + (size_t)0 * 16 * STRIDE + k, kb);
    const half16 an1 = load_a_frag(arow0 + (size_t)1 * 16 * STRIDE + k, kb);
    const half16 an2 = load_a_frag(arow0 + (size_t)2 * 16 * STRIDE + k, kb);
    const half16 an3 = load_a_frag(arow0 + (size_t)3 * 16 * STRIDE + k, kb);
    acc[0] = WMMA_F16(ac0, bc, acc[0]);
    acc[1] = WMMA_F16(ac1, bc, acc[1]);
    acc[2] = WMMA_F16(ac2, bc, acc[2]);
    acc[3] = WMMA_F16(ac3, bc, acc[3]);
    bc = bn; ac0 = an0; ac1 = an1; ac2 = an2; ac3 = an3;
  }
  acc[0] = WMMA_F16(ac0, bc, acc[0]);
  acc[1] = WMMA_F16(ac1, bc, acc[1]);
  acc[2] = WMMA_F16(ac2, bc, acc[2]);
  acc[3] = WMMA_F16(ac3, bc, acc[3]);
}

// One LSTM timestep for one 16-wide hidden slice (hblk).
// Block: 128 threads = 4 waves; wave w == gate w (i,f,o,c), computing its gate for
// ALL 64 batch rows (4 M-tiles) => each weight fragment loaded once per WG, reused
// by 4 WMMAs. Gates exchanged through LDS for the fused elementwise cell update.
template <int DIN>
__device__ __forceinline__ void lstm_step_body(
    const _Float16* __restrict__ xt,    // [BB, DIN] f16
    const _Float16* __restrict__ Wt,    // [GG, DIN+HH] f16 (pre-transposed [W_x;W_h])
    const float*    __restrict__ bias,  // [GG]
    const _Float16* __restrict__ Hin,   // [BB, HH] f16
    _Float16*       __restrict__ Hout,  // [BB, HH] f16 (ping-pong)
    float*          __restrict__ Cf,    // [BB, HH] f32 (in/out, column-private per WG)
    float*          __restrict__ outH,  // [BB, HH] f32 slice of layer_outputs
    _Float16*       __restrict__ outHh, // [BB, HH] f16 archive for next layer (nullable)
    int hblk,
    float (*gbuf)[BB][17])              // LDS gate-exchange buffer
{
  constexpr int K = DIN + HH;
  const int lane = threadIdx.x & 31;
  const int gate = threadIdx.x >> 5;     // wave index == gate (0=i,1=f,2=o,3=c)
  const int nl   = lane & 15;
  const int kb   = (lane < 16) ? 0 : 8;
  const int kg   = (lane < 16) ? 0 : 16;
  const int h0   = hblk << 4;
  const int ncol = h0 + nl;              // hidden index in [0,HH)

  const _Float16* brow = Wt + (size_t)(gate * HH + ncol) * K;  // per-lane weight row
  const _Float16* ax0  = xt  + (size_t)nl * DIN;               // A row for m-tile 0
  const _Float16* ah0  = Hin + (size_t)nl * HH;

  float8 acc[4];
  #pragma unroll
  for (int mt = 0; mt < 4; ++mt) acc[mt] = (float8){0.f,0.f,0.f,0.f,0.f,0.f,0.f,0.f};

  gemm_seg<DIN, DIN>(ax0, brow, kb, kg, acc);        // x_t  @ W_x part
  gemm_seg<HH,  HH >(ah0, brow + DIN, kb, kg, acc);  // H_{t-1} @ W_h part

  // Stage biased gate values in LDS (pad 16->17: lane halves differ by 8 rows =
  // 128 dwords = 0 mod 64 banks without padding).
  const float vb = bias[gate * HH + ncol];
  #pragma unroll
  for (int mt = 0; mt < 4; ++mt) {
    #pragma unroll
    for (int r = 0; r < 8; ++r) {
      const int m = mt * 16 + ((lane < 16) ? r : (8 + r));   // C/D layout row
      gbuf[gate][m][nl] = acc[mt][r] + vb;
    }
  }
  __syncthreads();

  // Fused elementwise LSTM update: 128 threads x 8 rows each over the 64x16 slice
  const int col = threadIdx.x & 15;
  const int r0  = threadIdx.x >> 4;                          // 0..7
  #pragma unroll
  for (int j = 0; j < 8; ++j) {
    const int m = r0 + (j << 3);                             // 0..63
    const size_t idx = (size_t)m * HH + (h0 + col);
    const float iv = sigmoidf_(gbuf[0][m][col]);
    const float fv = sigmoidf_(gbuf[1][m][col]);
    const float ov = sigmoidf_(gbuf[2][m][col]);
    const float cv = tanhf(gbuf[3][m][col]);
    const float Cn = fv * Cf[idx] + iv * cv;
    Cf[idx] = Cn;
    const float Hn = ov * tanhf(Cn);
    outH[idx] = Hn;
    Hout[idx] = (_Float16)Hn;
    if (outHh) outHh[idx] = (_Float16)Hn;
  }
}

// Fused dual-layer step: blocks 0..63 run layer 0 at time t, blocks 64..127 run
// layer 1 at time t-1 (software pipeline across layers -> 2x resident waves and
// half the kernel launches on the sequential recurrent chain).
__global__ __launch_bounds__(128) void lstm_dual_step_kernel(
    const _Float16* __restrict__ xt0, const _Float16* __restrict__ W0,
    const float* __restrict__ b0, const _Float16* __restrict__ Hin0,
    _Float16* __restrict__ Hout0, float* __restrict__ Cf0,
    float* __restrict__ out0, _Float16* __restrict__ arch0, int do0,
    const _Float16* __restrict__ xt1, const _Float16* __restrict__ W1,
    const float* __restrict__ b1, const _Float16* __restrict__ Hin1,
    _Float16* __restrict__ Hout1, float* __restrict__ Cf1,
    float* __restrict__ out1, int do1)
{
  __shared__ float gbuf[4][BB][17];
  if (blockIdx.x < (HH / 16)) {
    if (!do0) return;
    lstm_step_body<DD>(xt0, W0, b0, Hin0, Hout0, Cf0, out0, arch0,
                       (int)blockIdx.x, gbuf);
  } else {
    if (!do1) return;
    lstm_step_body<HH>(xt1, W1, b1, Hin1, Hout1, Cf1, out1, nullptr,
                       (int)blockIdx.x - (HH / 16), gbuf);
  }
}

__global__ void cvt_f32_f16_kernel(const float* __restrict__ src, _Float16* __restrict__ dst, long long n) {
  long long i = (long long)blockIdx.x * blockDim.x + threadIdx.x;
  if (i < n) dst[i] = (_Float16)src[i];
}

// Build Wt[n, k] = (k < Din ? Wx[k, n] : Wh[k-Din, n]) as f16, n in [0,GG), k in [0,Din+HH)
__global__ void build_wt_kernel(const float* __restrict__ Wx, const float* __restrict__ Wh,
                                int Din, _Float16* __restrict__ Wt, long long n) {
  long long i = (long long)blockIdx.x * blockDim.x + threadIdx.x;
  if (i >= n) return;
  const int K = Din + HH;
  const int row = (int)(i / K);
  const int k   = (int)(i % K);
  const float v = (k < Din) ? Wx[(size_t)k * GG + row] : Wh[(size_t)(k - Din) * GG + row];
  Wt[i] = (_Float16)v;
}

__global__ void zero_u32_kernel(unsigned int* __restrict__ p, long long n) {
  long long i = (long long)blockIdx.x * blockDim.x + threadIdx.x;
  if (i < n) p[i] = 0u;
}

__global__ void copy_f32_kernel(float* __restrict__ dst, const float* __restrict__ src, long long n) {
  long long i = (long long)blockIdx.x * blockDim.x + threadIdx.x;
  if (i < n) dst[i] = src[i];
}

extern "C" void kernel_launch(void* const* d_in, const int* in_sizes, int n_in,
                              void* d_out, int out_size, void* d_ws, size_t ws_size,
                              hipStream_t stream) {
  (void)in_sizes; (void)n_in; (void)out_size; (void)ws_size;

  const float* x   = (const float*)d_in[0];
  const float* Wx0 = (const float*)d_in[1];
  const float* Wh0 = (const float*)d_in[2];
  const float* b0  = (const float*)d_in[3];
  const float* Wx1 = (const float*)d_in[4];
  const float* Wh1 = (const float*)d_in[5];
  const float* b1  = (const float*)d_in[6];
  float* out = (float*)d_out;

  // Workspace carve-out (256B-aligned slices), ~131 MB total
  size_t off = 0;
  auto alloc = [&](size_t bytes) -> void* {
    void* p = (void*)((char*)d_ws + off);
    off += (bytes + 255) & ~(size_t)255;
    return p;
  };
  _Float16* x0h  = (_Float16*)alloc((size_t)TT * BB * DD * 2);       // x in f16
  _Float16* x1h  = (_Float16*)alloc((size_t)TT * BB * HH * 2);       // layer0 output in f16
  _Float16* W0t  = (_Float16*)alloc((size_t)GG * (DD + HH) * 2);     // [Wx0;Wh0]^T f16
  _Float16* W1t  = (_Float16*)alloc((size_t)GG * (HH + HH) * 2);     // [Wx1;Wh1]^T f16
  _Float16* Hh0a = (_Float16*)alloc((size_t)BB * HH * 2);            // layer0 H ping
  _Float16* Hh0b = (_Float16*)alloc((size_t)BB * HH * 2);            // layer0 H pong
  _Float16* Hh1a = (_Float16*)alloc((size_t)BB * HH * 2);            // layer1 H ping
  _Float16* Hh1b = (_Float16*)alloc((size_t)BB * HH * 2);            // layer1 H pong
  float*    Cf0  = (float*)   alloc((size_t)BB * HH * 4);            // layer0 C state
  float*    Cf1  = (float*)   alloc((size_t)BB * HH * 4);            // layer1 C state

  auto cdiv = [](long long a, long long b) -> long long { return (a + b - 1) / b; };

  // Precision conversion + weight transpose/concat
  {
    long long n = (long long)TT * BB * DD;
    cvt_f32_f16_kernel<<<(unsigned)cdiv(n, 256), 256, 0, stream>>>(x, x0h, n);
  }
  {
    long long n = (long long)GG * (DD + HH);
    build_wt_kernel<<<(unsigned)cdiv(n, 256), 256, 0, stream>>>(Wx0, Wh0, DD, W0t, n);
  }
  {
    long long n = (long long)GG * (HH + HH);
    build_wt_kernel<<<(unsigned)cdiv(n, 256), 256, 0, stream>>>(Wx1, Wh1, HH, W1t, n);
  }
  // Zero all recurrent state (both layers) once up front
  {
    long long nh = (long long)BB * HH / 2;  // dwords per f16 H buffer
    zero_u32_kernel<<<(unsigned)cdiv(nh, 256), 256, 0, stream>>>((unsigned int*)Hh0a, nh);
    zero_u32_kernel<<<(unsigned)cdiv(nh, 256), 256, 0, stream>>>((unsigned int*)Hh0b, nh);
    zero_u32_kernel<<<(unsigned)cdiv(nh, 256), 256, 0, stream>>>((unsigned int*)Hh1a, nh);
    zero_u32_kernel<<<(unsigned)cdiv(nh, 256), 256, 0, stream>>>((unsigned int*)Hh1b, nh);
    long long nc = (long long)BB * HH;      // dwords per f32 C buffer
    zero_u32_kernel<<<(unsigned)cdiv(nc, 256), 256, 0, stream>>>((unsigned int*)Cf0, nc);
    zero_u32_kernel<<<(unsigned)cdiv(nc, 256), 256, 0, stream>>>((unsigned int*)Cf1, nc);
  }

  const long long slice = (long long)BB * HH;

  // Layer-pipelined time loop: step s runs layer0@t=s and layer1@t=s-1.
  for (int s = 0; s <= TT; ++s) {
    const int do0 = (s < TT) ? 1 : 0;
    const int t1  = s - 1;
    const int do1 = (t1 >= 0) ? 1 : 0;
    const int tc0 = do0 ? s : (TT - 1);   // clamped (pointers unused when do==0)
    const int tc1 = do1 ? t1 : 0;
    const _Float16* Hin0  = (tc0 & 1) ? Hh0b : Hh0a;
    _Float16*       Hout0 = (tc0 & 1) ? Hh0a : Hh0b;
    const _Float16* Hin1  = (tc1 & 1) ? Hh1b : Hh1a;
    _Float16*       Hout1 = (tc1 & 1) ? Hh1a : Hh1b;
    lstm_dual_step_kernel<<<2 * (HH / 16), 128, 0, stream>>>(
        x0h + (size_t)tc0 * BB * DD, W0t, b0, Hin0, Hout0, Cf0,
        out + (size_t)tc0 * slice, x1h + (size_t)tc0 * slice, do0,
        x1h + (size_t)tc1 * slice, W1t, b1, Hin1, Hout1, Cf1,
        out + (size_t)(TT + tc1) * slice, do1);
  }

  // last_H = H at t=T-1 (copy from layer_outputs), last_C from persistent C buffers
  copy_f32_kernel<<<(unsigned)cdiv(slice, 256), 256, 0, stream>>>(
      out + (size_t)LL * TT * slice, out + (size_t)(TT - 1) * slice, slice);
  copy_f32_kernel<<<(unsigned)cdiv(slice, 256), 256, 0, stream>>>(
      out + (size_t)LL * TT * slice + slice, out + (size_t)(2 * TT - 1) * slice, slice);
  copy_f32_kernel<<<(unsigned)cdiv(slice, 256), 256, 0, stream>>>(
      out + (size_t)LL * TT * slice + (size_t)LL * slice, Cf0, slice);
  copy_f32_kernel<<<(unsigned)cdiv(slice, 256), 256, 0, stream>>>(
      out + (size_t)LL * TT * slice + (size_t)LL * slice + slice, Cf1, slice);
}
